// TemporalDeformableSelfAttentionEncoder_23244363006247
// MI455X (gfx1250) — compile-verified, matrix operand order checked by reading, not run on hardware
//
#include <hip/hip_runtime.h>

// ---- problem constants (from reference) ----
#define LQS   5440      // sum of level sizes
#define CCH   256       // channels
#define NHEAD 8
#define DHD   32        // head dim
#define NLV   4
#define NPT   4
#define DFFN  1024
#define BBB   2         // batch
#define NNN   3         // temporal slots

typedef __attribute__((ext_vector_type(16))) __bf16 v16bf;
typedef __attribute__((ext_vector_type(8)))  float  v8f;
typedef int v4i_gcc __attribute__((vector_size(16)));

#if defined(__has_builtin)
#if __has_builtin(__builtin_amdgcn_global_load_async_to_lds_b128) && \
    __has_builtin(__builtin_amdgcn_s_wait_asynccnt)
#define HAVE_ASYNC_LDS 1
#endif
#endif
#ifndef HAVE_ASYNC_LDS
#define HAVE_ASYNC_LDS 0
#endif

__device__ __forceinline__ unsigned short f2bfu(float f) {
  unsigned u = __builtin_bit_cast(unsigned, f);
  unsigned r = u + 0x7FFFu + ((u >> 16) & 1u);   // round-to-nearest-even
  return (unsigned short)(r >> 16);
}
__device__ __forceinline__ __bf16 f2bf(float f) {
  unsigned short s = f2bfu(f);
  return __builtin_bit_cast(__bf16, s);
}

#if HAVE_ASYNC_LDS
__device__ __forceinline__ void async_cp16(const __bf16* g, __bf16* l) {
  __builtin_amdgcn_global_load_async_to_lds_b128(
      (__attribute__((address_space(1))) v4i_gcc*)g,
      (__attribute__((address_space(3))) v4i_gcc*)l, 0, 0);
}
#endif

// ------------------------------------------------------------------
// WMMA GEMM: Out[M,N] = act(A[M,K] * Wt[N,K]^T + bias[N])
// A: bf16 [M,K] row-major (per-batch stride), Wt: bf16 [N,K] row-major
// (weights pre-transposed so BOTH tiles are contiguous row chunks ->
// async global->LDS 16B copies, no on-the-fly transpose).
// Block = 256 threads (8 waves). Block tile 64x128, wave tile 32x32
// (4 WMMAs / k-step / wave). Double-buffered LDS: the next tile's
// async loads fly behind the current tile's WMMAs; s_wait_asynccnt(3)
// waits only for the oldest 3 ops (current buffer).
// grid = (N/128, M/64, B). Optional packed-bf16 secondary output.
// ------------------------------------------------------------------
#define TSTR 48   // LDS row stride in bf16 elems (32 data + 16 pad = 96B)

__global__ __launch_bounds__(256)
void gemm_wmma_bf16(const __bf16* __restrict__ A, long strideA,
                    const __bf16* __restrict__ Wt,
                    const float*  __restrict__ bias,
                    float* __restrict__ Out, long strideC,
                    __bf16* __restrict__ OutBf, long strideCbf,
                    int M, int N, int K, int relu)
{
  __shared__ __align__(16) __bf16 As[2][64  * TSTR];   // [m][k]
  __shared__ __align__(16) __bf16 Bs[2][128 * TSTR];   // [n][k]

  A   += (long)blockIdx.z * strideA;
  Out += (long)blockIdx.z * strideC;
  if (OutBf) OutBf += (long)blockIdx.z * strideCbf;

  const int m0   = blockIdx.y * 64;
  const int n0   = blockIdx.x * 128;
  const int t    = threadIdx.x;
  const int lane = t & 31;
  const int wave = t >> 5;

  const int mpair = wave & 1;    // 0..1 -> 32-row half of the 64-row tile
  const int npair = wave >> 1;   // 0..3 -> 32-col slice of the 128-col tile
  const int half  = lane >> 4;   // K-half selector per 16-bit WMMA layout
  const int l16   = lane & 15;

  v8f acc[4];
  #pragma unroll
  for (int i = 0; i < 4; ++i) acc[i] = (v8f){0.f,0.f,0.f,0.f,0.f,0.f,0.f,0.f};

  // staging chunk coordinates (16B = 8 bf16 per chunk)
  const int ar = t >> 2, ac = t & 3;            // A: 64 rows x 4 chunks
  const int br0 = t >> 1, bc0 = (t & 1) * 2;    // B: 128 rows x 4 chunks, 2/thread

  const __bf16* gaBase = A  + (long)(m0 + ar)  * K + ac  * 8;
  const __bf16* gbBase = Wt + (long)(n0 + br0) * K + bc0 * 8;
  __bf16* const laA  = &As[0][ar  * TSTR + ac * 8];
  __bf16* const laB0 = &Bs[0][br0 * TSTR + bc0 * 8];
  const int bufStepA = 64 * TSTR, bufStepB = 128 * TSTR;

  auto issue = [&](int k0, int buf) {
#if HAVE_ASYNC_LDS
    async_cp16(gaBase + k0,     laA  + buf * bufStepA);
    async_cp16(gbBase + k0,     laB0 + buf * bufStepB);
    async_cp16(gbBase + k0 + 8, laB0 + buf * bufStepB + 8);
#else
    if (k0 + 32 < K) __builtin_prefetch(gaBase + k0 + 32, 0, 0);
    *(uint4*)(laA  + buf * bufStepA)     = *(const uint4*)(const void*)(gaBase + k0);
    *(uint4*)(laB0 + buf * bufStepB)     = *(const uint4*)(const void*)(gbBase + k0);
    *(uint4*)(laB0 + buf * bufStepB + 8) = *(const uint4*)(const void*)(gbBase + k0 + 8);
#endif
  };

  const int nIter = K >> 5;
  issue(0, 0);
  for (int it = 0; it < nIter; ++it) {
    const int cur = it & 1;
    if (it + 1 < nIter) {
      issue((it + 1) << 5, cur ^ 1);
#if HAVE_ASYNC_LDS
      __builtin_amdgcn_s_wait_asynccnt(3);   // oldest 3 (current buf) done
#endif
    } else {
#if HAVE_ASYNC_LDS
      __builtin_amdgcn_s_wait_asynccnt(0);
#endif
    }
    __syncthreads();

    // fragments: lane<16 -> K 0..7 & 16..23 ; lane>=16 -> K 8..15 & 24..31
    v16bf a0, a1, b0, b1;
    {
      const __bf16* p;
      p = &As[cur][(mpair * 32 +  0 + l16) * TSTR + half * 8];
      #pragma unroll
      for (int e = 0; e < 8; ++e) { a0[e] = p[e]; a0[8 + e] = p[16 + e]; }
      p = &As[cur][(mpair * 32 + 16 + l16) * TSTR + half * 8];
      #pragma unroll
      for (int e = 0; e < 8; ++e) { a1[e] = p[e]; a1[8 + e] = p[16 + e]; }
      p = &Bs[cur][(npair * 32 +  0 + l16) * TSTR + half * 8];
      #pragma unroll
      for (int e = 0; e < 8; ++e) { b0[e] = p[e]; b0[8 + e] = p[16 + e]; }
      p = &Bs[cur][(npair * 32 + 16 + l16) * TSTR + half * 8];
      #pragma unroll
      for (int e = 0; e < 8; ++e) { b1[e] = p[e]; b1[8 + e] = p[16 + e]; }
    }
    acc[0] = __builtin_amdgcn_wmma_f32_16x16x32_bf16(false, a0, false, b0, (short)0, acc[0], false, false);
    acc[1] = __builtin_amdgcn_wmma_f32_16x16x32_bf16(false, a0, false, b1, (short)0, acc[1], false, false);
    acc[2] = __builtin_amdgcn_wmma_f32_16x16x32_bf16(false, a1, false, b0, (short)0, acc[2], false, false);
    acc[3] = __builtin_amdgcn_wmma_f32_16x16x32_bf16(false, a1, false, b1, (short)0, acc[3], false, false);
    __syncthreads();
  }

  // writeback: C/D layout: lane col = l16, VGPR v -> row v + half*8
  #pragma unroll
  for (int i = 0; i < 2; ++i) {
    #pragma unroll
    for (int j = 0; j < 2; ++j) {
      const v8f a = acc[i * 2 + j];
      const int gm = m0 + (mpair * 2 + i) * 16;
      const int gn = n0 + (npair * 2 + j) * 16 + l16;
      const float bv = bias[gn];
      #pragma unroll
      for (int v = 0; v < 8; ++v) {
        const int m = gm + v + half * 8;
        float r = a[v] + bv;
        if (relu) r = r > 0.f ? r : 0.f;
        Out[(long)m * N + gn] = r;
        if (OutBf) OutBf[(long)m * N + gn] = f2bf(r);
      }
    }
  }
}

// ------------------------------------------------------------------
// f32 -> bf16 packed conversion (n multiple of 4)
// ------------------------------------------------------------------
__global__ __launch_bounds__(256)
void cvt_bf16_kernel(const float* __restrict__ s, __bf16* __restrict__ d, long n)
{
  const long i = (long)blockIdx.x * 256 + threadIdx.x;
  if (i * 4 >= n) return;
  const float4 v = ((const float4*)s)[i];
  ushort4 o;
  o.x = f2bfu(v.x); o.y = f2bfu(v.y); o.z = f2bfu(v.z); o.w = f2bfu(v.w);
  ((ushort4*)d)[i] = o;
}

// W[K,N] f32 -> Wt[N,K] bf16
__global__ __launch_bounds__(256)
void cvt_transpose_bf16(const float* __restrict__ W, __bf16* __restrict__ Wt,
                        int K, int N)
{
  const long i = (long)blockIdx.x * 256 + threadIdx.x;
  if (i >= (long)K * N) return;
  const int k = (int)(i / N);
  const int n = (int)(i - (long)k * N);
  Wt[(long)n * K + k] = f2bf(W[i]);
}

// ------------------------------------------------------------------
// Softmax over 16 (level,point) logits per (b, lq, head). In place.
// ------------------------------------------------------------------
__global__ __launch_bounds__(256)
void softmax16_kernel(float* __restrict__ aw)
{
  const int i = blockIdx.x * 256 + threadIdx.x;
  if (i >= BBB * LQS * NHEAD) return;
  float* p = aw + (long)i * 16;
  float m = p[0];
  #pragma unroll
  for (int j = 1; j < 16; ++j) m = fmaxf(m, p[j]);
  float e[16], s = 0.f;
  #pragma unroll
  for (int j = 0; j < 16; ++j) { e[j] = __expf(p[j] - m); s += e[j]; }
  const float inv = 1.f / s;
  #pragma unroll
  for (int j = 0; j < 16; ++j) p[j] = e[j] * inv;
}

// ------------------------------------------------------------------
// Multi-scale deformable attention sampling. thread = (b,lq,h,d).
// Emits f32 and packed-bf16 outputs (bf16 feeds the W_out GEMM).
// ------------------------------------------------------------------
__global__ __launch_bounds__(256)
void msdeform_kernel(const float* __restrict__ value,
                     const float* __restrict__ off,
                     const float* __restrict__ aw,
                     float* __restrict__ out,
                     __bf16* __restrict__ outbf)
{
  const long gid = (long)blockIdx.x * 256 + threadIdx.x;
  const int d    = (int)(gid & 31);
  const long rest = gid >> 5;                 // (b*LQ + lq)*NH + h
  if (rest >= (long)BBB * LQS * NHEAD) return;
  const int h    = (int)(rest & 7);
  const long row = rest >> 3;                 // b*LQ + lq
  const int b  = (int)(row / LQS);
  const int lq = (int)(row - (long)b * LQS);

  const int Hs[4]     = {64, 32, 16, 8};
  const int Ws[4]     = {64, 32, 16, 8};
  const int starts[4] = {0, 4096, 5120, 5376};

  float rx, ry;
  {
    const int lvl = (lq < 4096) ? 0 : (lq < 5120 ? 1 : (lq < 5376 ? 2 : 3));
    const int idx = lq - starts[lvl];
    const int W_  = Ws[lvl];
    const int y   = idx / W_;
    const int x   = idx - y * W_;
    rx = (x + 0.5f) / (float)W_;
    ry = (y + 0.5f) / (float)Hs[lvl];
  }

  const long offBase = (row * NHEAD + h) * (NLV * NPT * 2);
  const long awBase  = (row * NHEAD + h) * (NLV * NPT);
  const float* vb = value + (long)b * LQS * CCH;

  float accd = 0.f;
  #pragma unroll
  for (int lvl = 0; lvl < NLV; ++lvl) {
    const int H_ = Hs[lvl], W_ = Ws[lvl], st = starts[lvl];
    #pragma unroll
    for (int p = 0; p < NPT; ++p) {
      const float ox = off[offBase + (lvl * NPT + p) * 2 + 0];
      const float oy = off[offBase + (lvl * NPT + p) * 2 + 1];
      const float w  = aw[awBase + lvl * NPT + p];
      const float px = (rx + ox / (float)W_) * (float)W_ - 0.5f;
      const float py = (ry + oy / (float)H_) * (float)H_ - 0.5f;
      const float x0 = floorf(px), y0 = floorf(py);
      float s = 0.f;
      #pragma unroll
      for (int dy = 0; dy < 2; ++dy) {
        #pragma unroll
        for (int dx = 0; dx < 2; ++dx) {
          const float xi = x0 + (float)dx, yi = y0 + (float)dy;
          const float bw = (1.f - fabsf(px - xi)) * (1.f - fabsf(py - yi));
          const bool valid = (xi >= 0.f) && (xi <= (float)(W_ - 1)) &&
                             (yi >= 0.f) && (yi <= (float)(H_ - 1));
          const int xc = (int)fminf(fmaxf(xi, 0.f), (float)(W_ - 1));
          const int yc = (int)fminf(fmaxf(yi, 0.f), (float)(H_ - 1));
          const int pos = st + yc * W_ + xc;
          const float v = vb[(long)pos * CCH + h * DHD + d];
          s += valid ? v * bw : 0.f;
        }
      }
      accd += s * w;
    }
  }
  out[row * CCH + h * DHD + d]   = accd;
  outbf[row * CCH + h * DHD + d] = f2bf(accd);
}

// ------------------------------------------------------------------
// x = LayerNorm(x + res) * g + b, one wave32 per row of 256.
// Also writes packed bf16 copy of the row (next GEMM's A operand).
// ------------------------------------------------------------------
__global__ __launch_bounds__(256)
void add_ln_kernel(float* __restrict__ x, const float* __restrict__ res,
                   const float* __restrict__ g, const float* __restrict__ bta,
                   __bf16* __restrict__ xbf)
{
  const int wid  = (blockIdx.x * 256 + threadIdx.x) >> 5;
  const int lane = threadIdx.x & 31;
  if (wid >= BBB * LQS) return;
  float* xr = x + (long)wid * CCH;
  const float* rr = res + (long)wid * CCH;
  __bf16* hb = xbf + (long)wid * CCH;
  float v[8], s = 0.f, ss = 0.f;
  #pragma unroll
  for (int j = 0; j < 8; ++j) {
    const int c = j * 32 + lane;
    const float t = xr[c] + rr[c];
    v[j] = t; s += t; ss += t * t;
  }
  #pragma unroll
  for (int m = 16; m >= 1; m >>= 1) {
    s  += __shfl_xor(s,  m, 32);
    ss += __shfl_xor(ss, m, 32);
  }
  const float mean = s * (1.f / CCH);
  const float var  = ss * (1.f / CCH) - mean * mean;
  const float rstd = rsqrtf(var + 1e-5f);
  #pragma unroll
  for (int j = 0; j < 8; ++j) {
    const int c = j * 32 + lane;
    const float o = (v[j] - mean) * rstd * g[c] + bta[c];
    xr[c] = o;
    hb[c] = f2bf(o);
  }
}

// ------------------------------------------------------------------
// memory[:, q] -> att (f32 + bf16)   and   att -> out[:, q]
// ------------------------------------------------------------------
__global__ __launch_bounds__(256)
void gather_q_kernel(const float* __restrict__ mem, float* __restrict__ att,
                     __bf16* __restrict__ attbf, int q)
{
  const long per = (long)LQS * CCH / 4;
  const long i = (long)blockIdx.x * 256 + threadIdx.x;
  if (i >= BBB * per) return;
  const long b = i / per, rem = i - b * per;
  const float4 v = ((const float4*)mem)[(b * NNN + q) * per + rem];
  ((float4*)att)[b * per + rem] = v;
  ushort4 o;
  o.x = f2bfu(v.x); o.y = f2bfu(v.y); o.z = f2bfu(v.z); o.w = f2bfu(v.w);
  ((ushort4*)attbf)[b * per + rem] = o;
}

__global__ __launch_bounds__(256)
void scatter_q_kernel(const float* __restrict__ att, float* __restrict__ out, int q)
{
  const long per = (long)LQS * CCH / 4;
  const long i = (long)blockIdx.x * 256 + threadIdx.x;
  if (i >= BBB * per) return;
  const long b = i / per, rem = i - b * per;
  ((float4*)out)[(b * NNN + q) * per + rem] = ((const float4*)att)[b * per + rem];
}

// ------------------------------------------------------------------
extern "C" void kernel_launch(void* const* d_in, const int* in_sizes, int n_in,
                              void* d_out, int out_size, void* d_ws, size_t ws_size,
                              hipStream_t stream)
{
  (void)in_sizes; (void)n_in; (void)out_size; (void)ws_size;
  const float* memory = (const float*)d_in[0];
  const float* W_off  = (const float*)d_in[1];
  const float* b_off  = (const float*)d_in[2];
  const float* W_attw = (const float*)d_in[3];
  const float* b_attw = (const float*)d_in[4];
  const float* W_val  = (const float*)d_in[5];
  const float* b_val  = (const float*)d_in[6];
  const float* W_out  = (const float*)d_in[7];
  const float* b_out  = (const float*)d_in[8];
  const float* ln1_g  = (const float*)d_in[9];
  const float* ln1_b  = (const float*)d_in[10];
  const float* W_ff1  = (const float*)d_in[11];
  const float* b_ff1  = (const float*)d_in[12];
  const float* W_ff2  = (const float*)d_in[13];
  const float* b_ff2  = (const float*)d_in[14];
  const float* ln2_g  = (const float*)d_in[15];
  const float* ln2_b  = (const float*)d_in[16];

  // ---- scratch layout ----
  const long RC  = (long)BBB * LQS * CCH;           // 2,785,280
  const long RH  = (long)BBB * LQS * DFFN;
  const long MEMN = (long)BBB * NNN * LQS * CCH;
  char* p = (char*)d_ws;
  auto alloc_f = [&](long n) { float*  r = (float*)p;  p += n * sizeof(float);  return r; };
  auto alloc_h = [&](long n) { __bf16* r = (__bf16*)p; p += n * sizeof(__bf16); if ((size_t)p & 15) p += 16 - ((size_t)p & 15); return r; };

  float* att   = alloc_f(RC);
  float* val   = alloc_f(RC);
  float* offb  = alloc_f(RC);
  float* awb   = alloc_f((long)BBB * LQS * 128);
  float* samp  = alloc_f(RC);
  float* tmp   = alloc_f(RC);
  float* hid   = alloc_f(RH);
  __bf16* mem_bf  = alloc_h(MEMN);
  __bf16* att_bf  = alloc_h(RC);
  __bf16* samp_bf = alloc_h(RC);
  __bf16* hid_bf  = alloc_h(RH);
  __bf16* wv_t[2], *woff_t[2], *wa_t[2], *wout_t[2], *wf1_t[2], *wf2_t[2];
  for (int l = 0; l < 2; ++l) {
    wv_t[l]   = alloc_h(256 * 256);
    woff_t[l] = alloc_h(256 * 256);
    wa_t[l]   = alloc_h(128 * 256);
    wout_t[l] = alloc_h(256 * 256);
    wf1_t[l]  = alloc_h(1024 * 256);
    wf2_t[l]  = alloc_h(256 * 1024);
  }

  const dim3 blk(256);
  auto cvtT = [&](const float* W, __bf16* Wt, int K, int N) {
    const long n = (long)K * N;
    cvt_transpose_bf16<<<(unsigned)((n + 255) / 256), blk, 0, stream>>>(W, Wt, K, N);
  };
  // one-time (per launch) operand preparation
  cvt_bf16_kernel<<<(unsigned)((MEMN / 4 + 255) / 256), blk, 0, stream>>>(memory, mem_bf, MEMN);
  for (int l = 0; l < 2; ++l) {
    cvtT(W_val  + (long)l * 256 * 256,  wv_t[l],   256, 256);
    cvtT(W_off  + (long)l * 256 * 256,  woff_t[l], 256, 256);
    cvtT(W_attw + (long)l * 256 * 128,  wa_t[l],   256, 128);
    cvtT(W_out  + (long)l * 256 * 256,  wout_t[l], 256, 256);
    cvtT(W_ff1  + (long)l * 256 * 1024, wf1_t[l],  256, 1024);
    cvtT(W_ff2  + (long)l * 1024 * 256, wf2_t[l],  1024, 256);
  }

  auto gemm = [&](const __bf16* A, long sA, const __bf16* Wt, const float* bias,
                  float* O, long sO, __bf16* Obf, long sObf,
                  int M, int N, int K, int relu) {
    dim3 grid(N / 128, M / 64, BBB);
    hipLaunchKernelGGL(gemm_wmma_bf16, grid, blk, 0, stream,
                       A, sA, Wt, bias, O, sO, Obf, sObf, M, N, K, relu);
  };

  const long cp4 = (long)BBB * LQS * CCH / 4;
  const int  smN = BBB * LQS * NHEAD;
  const long dfN = (long)BBB * LQS * NHEAD * 32;
  const int  lnB = (int)((BBB * LQS * 32 + 255) / 256);

  for (int q = 0; q < NNN; ++q) {
    gather_q_kernel<<<(unsigned)((cp4 + 255) / 256), blk, 0, stream>>>(memory, att, att_bf, q);
    for (int kvi = 0; kvi < NNN; ++kvi) {
      if (kvi == q) continue;
      for (int l = 0; l < 2; ++l) {
        // value = kv @ W_val + b_val
        gemm(mem_bf + (long)kvi * LQS * CCH, (long)NNN * LQS * CCH,
             wv_t[l], b_val + (long)l * CCH,
             val, (long)LQS * CCH, nullptr, 0, LQS, CCH, CCH, 0);
        // off = att @ W_off + b_off
        gemm(att_bf, (long)LQS * CCH, woff_t[l], b_off + (long)l * 256,
             offb, (long)LQS * 256, nullptr, 0, LQS, 256, CCH, 0);
        // aw = softmax(att @ W_attw + b_attw)
        gemm(att_bf, (long)LQS * CCH, wa_t[l], b_attw + (long)l * 128,
             awb, (long)LQS * 128, nullptr, 0, LQS, 128, CCH, 0);
        softmax16_kernel<<<(unsigned)((smN + 255) / 256), blk, 0, stream>>>(awb);
        // multi-scale deformable sampling
        msdeform_kernel<<<(unsigned)((dfN + 255) / 256), blk, 0, stream>>>(val, offb, awb, samp, samp_bf);
        // attn = samp @ W_out + b_out ; att = LN(att + attn)
        gemm(samp_bf, (long)LQS * CCH, wout_t[l], b_out + (long)l * CCH,
             tmp, (long)LQS * CCH, nullptr, 0, LQS, CCH, CCH, 0);
        add_ln_kernel<<<lnB, blk, 0, stream>>>(att, tmp,
             ln1_g + (long)l * CCH, ln1_b + (long)l * CCH, att_bf);
        // ffn
        gemm(att_bf, (long)LQS * CCH, wf1_t[l], b_ff1 + (long)l * DFFN,
             hid, (long)LQS * DFFN, hid_bf, (long)LQS * DFFN, LQS, DFFN, CCH, 1);
        gemm(hid_bf, (long)LQS * DFFN, wf2_t[l], b_ff2 + (long)l * CCH,
             tmp, (long)LQS * CCH, nullptr, 0, LQS, CCH, DFFN, 0);
        add_ln_kernel<<<lnB, blk, 0, stream>>>(att, tmp,
             ln2_g + (long)l * CCH, ln2_b + (long)l * CCH, att_bf);
      }
    }
    scatter_q_kernel<<<(unsigned)((cp4 + 255) / 256), blk, 0, stream>>>(att, (float*)d_out, q);
  }
}
